// MultiHeadAttention_1700807049639
// MI455X (gfx1250) — matrix operand order repaired and checked by reading, not compiled, not run on hardware
//
#include <hip/hip_runtime.h>

#define DEV_INLINE __device__ __forceinline__

constexpr int Bsz = 2, Ssz = 2048, Dsz = 1024, Hsz = 16, DKsz = 64;

typedef __attribute__((ext_vector_type(16))) __bf16        v16bf;
typedef __attribute__((ext_vector_type(8)))  float         v8f;
typedef __attribute__((ext_vector_type(8)))  unsigned int  v8u;
typedef __attribute__((ext_vector_type(4)))  unsigned int  v4u;
typedef __attribute__((ext_vector_type(8)))  int           v8i;
typedef __attribute__((ext_vector_type(4)))  int           v4i;

// ---------- helpers ----------------------------------------------------------

DEV_INLINE unsigned short f2bf(float f) {           // fp32 -> bf16 bits, RNE
  unsigned int u = __builtin_bit_cast(unsigned int, f);
  u += 0x7FFFu + ((u >> 16) & 1u);
  return (unsigned short)(u >> 16);
}

DEV_INLINE v8f wmma_bf16(v8u a, v8u b, v8f c) {
  return __builtin_amdgcn_wmma_f32_16x16x32_bf16(
      false, __builtin_bit_cast(v16bf, a),
      false, __builtin_bit_cast(v16bf, b),
      (short)0, c, false, false);
}

// 16-lane butterfly reduction via DPP (stays within DPP16 rows -> halves of the
// wave reduce independently, matching the C-layout row assignment). Pure VALU.
template <int CTRL>
DEV_INLINE float dpp_bfly(float x) {
  const int xi = __builtin_bit_cast(int, x);
  return __builtin_bit_cast(float,
      __builtin_amdgcn_update_dpp(xi, xi, CTRL, 0xF, 0xF, false));
}
DEV_INLINE float redmax16(float x) {
  x = fmaxf(x, dpp_bfly<0x0B1>(x));   // quad_perm(1,0,3,2)
  x = fmaxf(x, dpp_bfly<0x04E>(x));   // quad_perm(2,3,0,1)
  x = fmaxf(x, dpp_bfly<0x141>(x));   // row_half_mirror
  x = fmaxf(x, dpp_bfly<0x140>(x));   // row_mirror
  return x;
}

// ---------- GEMM: C[M,N] = X[M,K] * W[N,K]^T  (bf16 WMMA, fp32 acc) ----------

constexpr int TK  = 64;
constexpr int LDA = TK + 2;      // 33-word row stride, gcd(33,64)=1 -> conflict-free

DEV_INLINE void cvt4_to_lds(unsigned short* dst, const float* src) {
  float4 f = *(const float4*)src;
  *(unsigned int*)(dst)     = (unsigned int)f2bf(f.x) | ((unsigned int)f2bf(f.y) << 16);
  *(unsigned int*)(dst + 2) = (unsigned int)f2bf(f.z) | ((unsigned int)f2bf(f.w) << 16);
}
DEV_INLINE void cvt4_to_lds(unsigned short* dst, const unsigned short* src) {
  uint2 u = *(const uint2*)src;                      // already bf16 bits
  *(unsigned int*)(dst)     = u.x;
  *(unsigned int*)(dst + 2) = u.y;
}

DEV_INLINE void store_out(float* p, float v)          { *p = v; }
DEV_INLINE void store_out(unsigned short* p, float v) { *p = f2bf(v); }

// A fragment: 16x32 bf16, ISA 7.12.2 (lane m=l&15; v<4: K=h*8+2v; v>=4: K=16+h*8+2(v-4))
DEV_INLINE v8u frag_a_lds(const unsigned short* base, int koff, int lane) {
  const int m = lane & 15, h = lane >> 4;
  v8u a;
#pragma unroll
  for (int v = 0; v < 8; ++v) {
    const int k = koff + ((v < 4) ? (h * 8 + 2 * v) : (16 + h * 8 + 2 * (v - 4)));
    a[v] = *(const unsigned int*)(base + m * LDA + k);
  }
  return a;
}
// B fragment: 32x16 bf16 (lane n=l&15 holds column n; v<4: K=h*8+2v; v>=4: K=16+h*8+2(v-4))
DEV_INLINE v8u frag_b_lds(const unsigned short* base, int koff, int lane) {
  const int n = lane & 15, h = lane >> 4;
  v8u b;
#pragma unroll
  for (int v = 0; v < 8; ++v) {
    const int k = koff + ((v < 4) ? (h * 8 + 2 * v) : (16 + h * 8 + 2 * (v - 4)));
    b[v] = *(const unsigned int*)(base + n * LDA + k);
  }
  return b;
}

template <typename TIN, typename TOUT>
__global__ __launch_bounds__(256)
void gemm_xwt_bf16(const TIN* __restrict__ X, const float* __restrict__ W,
                   TOUT* __restrict__ C, int M, int N, int K) {
  __shared__ unsigned short As[128 * LDA];
  __shared__ unsigned short Bs[64 * LDA];

  const int tid = threadIdx.x, lane = tid & 31, wid = tid >> 5;
  const int wm = wid & 3, wn = wid >> 2;             // 4 (M) x 2 (N) waves
  const int m0 = blockIdx.x * 128, n0 = blockIdx.y * 64;

  v8f acc[2][2] = {};

  for (int k0 = 0; k0 < K; k0 += TK) {
    for (int i = tid; i < 128 * TK / 4; i += 256) {  // A tile: 128x64
      const int r = i >> 4, c = (i & 15) * 4;
      cvt4_to_lds(&As[r * LDA + c], X + (size_t)(m0 + r) * K + k0 + c);
    }
    for (int i = tid; i < 64 * TK / 4; i += 256) {   // B tile: 64x64 (weights)
      const int r = i >> 4, c = (i & 15) * 4;
      cvt4_to_lds(&Bs[r * LDA + c], W + (size_t)(n0 + r) * K + k0 + c);
    }
    __syncthreads();

#pragma unroll
    for (int kk = 0; kk < TK; kk += 32) {
      const v8u a0 = frag_a_lds(&As[(wm * 32)      * LDA], kk, lane);
      const v8u a1 = frag_a_lds(&As[(wm * 32 + 16) * LDA], kk, lane);
      const v8u b0 = frag_b_lds(&Bs[(wn * 32)      * LDA], kk, lane);
      const v8u b1 = frag_b_lds(&Bs[(wn * 32 + 16) * LDA], kk, lane);
      acc[0][0] = wmma_bf16(a0, b0, acc[0][0]);
      acc[0][1] = wmma_bf16(a0, b1, acc[0][1]);
      acc[1][0] = wmma_bf16(a1, b0, acc[1][0]);
      acc[1][1] = wmma_bf16(a1, b1, acc[1][1]);
    }
    __syncthreads();
  }

  const int n = lane & 15, h = lane >> 4;
#pragma unroll
  for (int i = 0; i < 2; ++i)
#pragma unroll
    for (int j = 0; j < 2; ++j)
#pragma unroll
      for (int v = 0; v < 8; ++v) {
        const int gm = m0 + wm * 32 + i * 16 + v + 8 * h;
        const int gn = n0 + wn * 32 + j * 16 + n;
        store_out(&C[(size_t)gm * N + gn], acc[i][j][v]);
      }
}

// ---------- TDM: async 2D tile (32 rows x 64 bf16) global -> padded LDS ------

constexpr int LDK = DKsz + 4;    // 68 elems = 32+2 dwords/row (TDM pads 2 dw / 32 dw)

DEV_INLINE void tdm_load_chunk(unsigned lds_byte_off, const unsigned short* gptr) {
  const unsigned long long ga = (unsigned long long)(size_t)gptr;
  v4u g0;
  g0[0] = 1u;                                        // count=1 (valid user D#)
  g0[1] = lds_byte_off;                              // lds_addr
  g0[2] = (unsigned)(ga & 0xFFFFFFFFull);            // global_addr[31:0]
  g0[3] = (unsigned)((ga >> 32) & 0x01FFFFFFull)     // global_addr[56:32]
        | (2u << 30);                                // type=2 ("image")
  v8i g1;
  g1[0] = (int)((1u << 16)                           // data_size = 2 bytes
              | (1u << 20)                           // pad_enable
              | (4u << 22)                           // pad_interval: 32 dwords
              | (1u << 25));                         // pad_amount:   2 dwords
  g1[1] = (int)(64u << 16);                          // tensor_dim0 = 64 (lo16)
  g1[2] = (int)(32u << 16);                          // tensor_dim0 hi=0 | tensor_dim1 = 32
  g1[3] = (int)(64u << 16);                          // tensor_dim1 hi=0 | tile_dim0 = 64
  g1[4] = 32;                                        // tile_dim1 = 32, tile_dim2 = 0
  g1[5] = (int)Dsz;                                  // tensor_dim0_stride = 1024 elems
  g1[6] = 0;                                         // stride hi | tensor_dim1_stride lo
  g1[7] = 0;
  const v4i z = {0, 0, 0, 0};
#if defined(__clang_major__) && __clang_major__ >= 23
  const v8i z8 = {0, 0, 0, 0, 0, 0, 0, 0};
  __builtin_amdgcn_tensor_load_to_lds(g0, g1, z, z, z8, 0);
#else
  __builtin_amdgcn_tensor_load_to_lds(g0, g1, z, z, 0);
#endif
}

// ---------- Flash attention: per-(b,h), 128 q-rows/block, 32-key chunks ------

__global__ __launch_bounds__(256)
void flash_attn_bf16(const unsigned short* __restrict__ Qb,
                     const unsigned short* __restrict__ Kb,
                     const unsigned short* __restrict__ Vb,
                     unsigned short* __restrict__ Ctx) {
  constexpr int LDP = 32 + 2;                        // 17-word stride, conflict-free
  __shared__ unsigned short Ks[2][32 * LDK];         // double-buffered TDM targets
  __shared__ unsigned short Vs[2][32 * LDK];
  __shared__ unsigned short Ps[8][16 * LDP];         // per-wave P transpose scratch

  const int tid = threadIdx.x, lane = tid & 31, wid = tid >> 5;
  const int b = blockIdx.z, h = blockIdx.y;
  const int q0 = blockIdx.x * 128 + wid * 16;        // this wave's 16 query rows
  const size_t base = (size_t)b * Ssz * Dsz + (size_t)h * DKsz;
  const unsigned short* qb = Qb + base;
  const unsigned short* kb = Kb + base;
  const unsigned short* vb = Vb + base;
  unsigned short*       cb = Ctx + base;

  const int ln = lane & 15, hl = lane >> 4;

  // Q fragments (A-layout, dk-tiles 0-31 / 32-63), resident for whole block
  v8u qf[2];
#pragma unroll
  for (int kt = 0; kt < 2; ++kt)
#pragma unroll
    for (int v = 0; v < 8; ++v) {
      const int dk = kt * 32 + ((v < 4) ? (hl * 8 + 2 * v) : (16 + hl * 8 + 2 * (v - 4)));
      qf[kt][v] = *(const unsigned int*)(qb + (size_t)(q0 + ln) * Dsz + dk);
    }

  // all-ones bf16 B fragment: row sums of P via one extra WMMA (P . 1)
  const v8u onesf = {0x3F803F80u, 0x3F803F80u, 0x3F803F80u, 0x3F803F80u,
                     0x3F803F80u, 0x3F803F80u, 0x3F803F80u, 0x3F803F80u};

  float mrow[8], lrow[8];
#pragma unroll
  for (int v = 0; v < 8; ++v) { mrow[v] = -3.0e38f; lrow[v] = 0.0f; }
  v8f accO[4] = {};

  const float scale = 0.125f;                        // 1/sqrt(64)
  const int nchunk = Ssz / 32;                       // 64

  // prologue: wave 0 kicks the TDM for chunk 0 into buffer 0
  if (wid == 0) {
    tdm_load_chunk((unsigned)(size_t)&Ks[0][0], kb);
    tdm_load_chunk((unsigned)(size_t)&Vs[0][0], vb);
  }

  for (int c = 0; c < nchunk; ++c) {
    const int cur = c & 1;
    if (wid == 0) {
      if (c + 1 < nchunk) {                          // prefetch next chunk, then wait
        const size_t off = (size_t)(c + 1) * 32 * Dsz;
        tdm_load_chunk((unsigned)(size_t)&Ks[cur ^ 1][0], kb + off);
        tdm_load_chunk((unsigned)(size_t)&Vs[cur ^ 1][0], vb + off);
        __builtin_amdgcn_s_wait_tensorcnt(2);        // in-order: current chunk done
      } else {
        __builtin_amdgcn_s_wait_tensorcnt(0);
      }
    }
    __syncthreads();                                 // buffer `cur` ready for all waves

    // scores: S = Q (16x64) * K^T (64x32) -> two 16x16 f32 tiles
    v8f sc[2] = {};
#pragma unroll
    for (int nt = 0; nt < 2; ++nt) {
#pragma unroll
      for (int kt = 0; kt < 2; ++kt) {
        v8u kf;                                      // B-layout: lane holds key column
#pragma unroll
        for (int v = 0; v < 8; ++v) {
          const int dk  = kt * 32 + ((v < 4) ? (hl * 8 + 2 * v) : (16 + hl * 8 + 2 * (v - 4)));
          const int key = nt * 16 + ln;
          kf[v] = *(const unsigned int*)&Ks[cur][key * LDK + dk];
        }
        sc[nt] = wmma_bf16(qf[kt], kf, sc[nt]);
      }
    }

    // online softmax; VGPR v of half hl holds row (v + 8*hl) across 16 lanes
    float corr[8];
#pragma unroll
    for (int v = 0; v < 8; ++v) {
      const float s0 = sc[0][v] * scale;
      const float s1 = sc[1][v] * scale;
      const float cm = redmax16(fmaxf(s0, s1));      // DPP butterfly, no DS traffic
      const float mn = fmaxf(mrow[v], cm);
      corr[v] = __expf(mrow[v] - mn);
      mrow[v] = mn;
      const float p0 = __expf(s0 - mn);
      const float p1 = __expf(s1 - mn);
      const int mr = v + 8 * hl;
      Ps[wid][mr * LDP + ln]      = f2bf(p0);        // C-layout -> A-layout via LDS
      Ps[wid][mr * LDP + 16 + ln] = f2bf(p1);
    }

    // wave-local LDS RAW: DS ops are in-order per wave; just drain DScnt
    asm volatile("s_wait_dscnt 0" ::: "memory");

    v8u pf;                                          // P as A-fragment (16 rows x 32 keys)
#pragma unroll
    for (int v = 0; v < 8; ++v) {
      const int k = (v < 4) ? (hl * 8 + 2 * v) : (16 + hl * 8 + 2 * (v - 4));
      pf[v] = *(const unsigned int*)&Ps[wid][ln * LDP + k];
    }

    // row sums of P with one WMMA against all-ones (lands directly in row layout)
    const v8f rsum = wmma_bf16(pf, onesf, v8f{});
#pragma unroll
    for (int v = 0; v < 8; ++v) lrow[v] = lrow[v] * corr[v] + rsum[v];

    // context: O = O*corr + P (16x32) * V (32x64)
#pragma unroll
    for (int t = 0; t < 4; ++t) {
#pragma unroll
      for (int v = 0; v < 8; ++v) accO[t][v] *= corr[v];
      v8u vf;                                        // B-layout: lane holds dk column
#pragma unroll
      for (int v = 0; v < 8; ++v) {
        const int k   = (v < 4) ? (hl * 8 + 2 * v) : (16 + hl * 8 + 2 * (v - 4));
        const int col = t * 16 + ln;
        const unsigned int lo = Vs[cur][k * LDK + col];
        const unsigned int hi = Vs[cur][(k + 1) * LDK + col];
        vf[v] = lo | (hi << 16);
      }
      accO[t] = wmma_bf16(pf, vf, accO[t]);
    }
    __syncthreads();                                 // all waves done with buffer `cur`
  }

  // normalize + store context (bf16) in [B,S,D] layout
#pragma unroll
  for (int t = 0; t < 4; ++t)
#pragma unroll
    for (int v = 0; v < 8; ++v) {
      const int mr = v + 8 * hl;
      const float o = accO[t][v] / lrow[v];
      cb[(size_t)(q0 + mr) * Dsz + t * 16 + ln] = f2bf(o);
    }
}

// ---------- launch -----------------------------------------------------------

extern "C" void kernel_launch(void* const* d_in, const int* in_sizes, int n_in,
                              void* d_out, int out_size, void* d_ws, size_t ws_size,
                              hipStream_t stream) {
  (void)in_sizes; (void)n_in; (void)out_size; (void)ws_size;

  const float* q  = (const float*)d_in[0];
  const float* k  = (const float*)d_in[1];
  const float* v  = (const float*)d_in[2];
  const float* wq = (const float*)d_in[3];
  const float* wk = (const float*)d_in[4];
  const float* wv = (const float*)d_in[5];
  const float* wo = (const float*)d_in[6];
  float* out = (float*)d_out;

  const size_t elems = (size_t)Bsz * Ssz * Dsz;      // 4 Mi bf16 elems per buffer
  unsigned short* Qb = (unsigned short*)d_ws;
  unsigned short* Kb = Qb + elems;
  unsigned short* Vb = Kb + elems;
  unsigned short* Cb = Vb + elems;                   // 32 MB total workspace

  const int M = Bsz * Ssz;                           // 4096
  const dim3 gemm_grid(M / 128, Dsz / 64);           // 32 x 16
  const dim3 attn_grid(Ssz / 128, Hsz, Bsz);         // 16 x 16 x 2

  gemm_xwt_bf16<float, unsigned short><<<gemm_grid, 256, 0, stream>>>(q, wq, Qb, M, Dsz, Dsz);
  gemm_xwt_bf16<float, unsigned short><<<gemm_grid, 256, 0, stream>>>(k, wk, Kb, M, Dsz, Dsz);
  gemm_xwt_bf16<float, unsigned short><<<gemm_grid, 256, 0, stream>>>(v, wv, Vb, M, Dsz, Dsz);
  flash_attn_bf16<<<attn_grid, 256, 0, stream>>>(Qb, Kb, Vb, Cb);
  gemm_xwt_bf16<unsigned short, float><<<gemm_grid, 256, 0, stream>>>(Cb, wo, out, M, Dsz, Dsz);
}
